// LSTM_25185688224139
// MI455X (gfx1250) — matrix operand order repaired and checked by reading, not compiled, not run on hardware
//
#include <hip/hip_runtime.h>
#include <hip/hip_bf16.h>
#include <stdint.h>

// ---------------------------------------------------------------------------
// LSTM forward on gfx1250 (MI455X), wave32 + WMMA bf16 (f32 accumulate).
//
//  * batch-split persistent scan: 8 blocks x 1024 threads, block b owns batch
//    rows [16b,16b+16). No cross-block sync (recurrence independent per row).
//  * gate preact = [x_t , h_{t-1}] @ [Wx ; Wh]^T : fused K=768 WMMA chain,
//    x_t staged to LDS each step, h double-buffered in LDS as bf16.
//  * cell state c: 16x16 f32 tile per wave, register-resident for all 512 t.
//  * weights pre-packed into a SPLIT-HALF B-fragment layout ([kb][half][lane]
//    16B planes): both b128 halves of a fragment load from one per-lane base
//    register with pure immediate offsets -> no per-iteration address math,
//    no xcnt chains (round 5). Bases laundered as addrspace(1) ints so loads
//    stay in-loop as global_load_b128 (no LICM->scratch spill, no flat).
//  * accumulators zero-init (inline-0 C); biases added post-GEMM.
//  * projection + row softmax on waves 0..15 (shfl_xor reductions).
// ---------------------------------------------------------------------------

typedef __attribute__((ext_vector_type(16))) __bf16    v16bf;
typedef __attribute__((ext_vector_type(8)))  float     v8f;
typedef __attribute__((ext_vector_type(4)))  unsigned  u32x4;   // POD 128-bit word
typedef __attribute__((address_space(1))) const u32x4  GQ4;     // global b128 word

#define B_   128
#define T_   512
#define I_   256
#define H_   512
#define O_   256
#define KTOT (I_ + H_)        // 768 : fused K for gate GEMMs
#define KBX  (I_ / 32)        // 8   : K-blocks from x
#define KBH  (H_ / 32)        // 16  : K-blocks from h
#define KBG  (KTOT / 32)      // 24
#define ROWX (I_ + 8)         // padded LDS row (bf16 elems)
#define ROWH (H_ + 8)         // padded LDS row (bf16 elems)

// ---------------- fragment loaders ----------------
// A fragment (16x32 bf16, ISA layout): lanes 0-15: M=lane, K = kb*32 + {0..7, 16..23}
//                                      lanes 16-31: M=lane-16, K = kb*32 + {8..15, 24..31}
__device__ __forceinline__ v16bf load_afrag(const __bf16* base, int rowStride, int kb) {
    const int lane = threadIdx.x & 31;
    const int m    = lane & 15;
    const int k0   = kb * 32 + ((lane & 16) ? 8 : 0);
    union { v16bf v; u32x4 q[2]; } f;
    const char* p = (const char*)(base + m * rowStride + k0);
    f.q[0] = *(const u32x4*)(p);
    f.q[1] = *(const u32x4*)(p + 32);   // +16 bf16 elements
    return f.v;
}

// B fragment, split-half layout: per k-block the 32 lanes' fragments occupy
// 1024B = two 512B planes; lane reads 16B from each plane at imm offsets.
__device__ __forceinline__ v16bf load_bfrag(uintptr_t base, int byteoff) {
    union { v16bf v; u32x4 q[2]; } f;
    f.q[0] = *(GQ4*)(base + byteoff);         // half 0 plane
    f.q[1] = *(GQ4*)(base + byteoff + 512);   // half 1 plane
    return f.v;
}

__device__ __forceinline__ float sigf(float x)  { return 1.0f / (1.0f + __expf(-x)); }
__device__ __forceinline__ float tanf_(float x) { return 1.0f - 2.0f / (__expf(2.0f * x) + 1.0f); }

#define WMMA_BF16(A, Bm, C) \
    __builtin_amdgcn_wmma_f32_16x16x32_bf16(false, (A), false, (Bm), (short)0, (C), false, false)

// Launder an address through inline asm so the optimizer cannot treat loads
// through it as loop-invariant (prevents hoist-and-spill-to-scratch).
#define LAUNDER_ADDR(a) asm volatile("" : "+v"(a))

// ---------------- prep kernels ----------------
__global__ void k_cvt_x(const float* __restrict__ x, __bf16* __restrict__ dst, int n) {
    int i = blockIdx.x * blockDim.x + threadIdx.x;
    if (i < n) dst[i] = (__bf16)x[i];
}

// Pack fused-gate weights [Wx(HxI) ; Wh(HxH)] into split-half fragment order:
// dst[(nt*KBG + kb)*512 + half*256 + lane*8 + jl] = W[nt*16 + (lane&15)][kb*32 + half*16 + (lane>>4)*8 + jl]
__global__ void k_pack_gate(const float* __restrict__ Wx, const float* __restrict__ Wh,
                            __bf16* __restrict__ dst) {
    int idx = blockIdx.x * blockDim.x + threadIdx.x;
    const int total = H_ * KTOT;
    if (idx >= total) return;
    int jl   = idx & 7;
    int lane = (idx >> 3) & 31;
    int half = (idx >> 8) & 1;
    int kb   = (idx >> 9) % KBG;
    int nt   = idx / (KBG * 512);
    int n = nt * 16 + (lane & 15);
    int k = kb * 32 + half * 16 + (lane >> 4) * 8 + jl;
    float v = (k < I_) ? Wx[n * I_ + k] : Wh[n * H_ + (k - I_)];
    dst[idx] = (__bf16)v;
}

// Pack projection weights Wph (O x H) the same way (KB = KBH).
__global__ void k_pack_proj(const float* __restrict__ W, __bf16* __restrict__ dst) {
    int idx = blockIdx.x * blockDim.x + threadIdx.x;
    const int total = O_ * H_;
    if (idx >= total) return;
    int jl   = idx & 7;
    int lane = (idx >> 3) & 31;
    int half = (idx >> 8) & 1;
    int kb   = (idx >> 9) & (KBH - 1);
    int nt   = idx / (KBH * 512);
    dst[idx] = (__bf16)W[(nt * 16 + (lane & 15)) * H_ + kb * 32 + half * 16 + (lane >> 4) * 8 + jl];
}

__global__ void k_add3(const float* __restrict__ a, const float* __restrict__ b,
                       const float* __restrict__ c, float* __restrict__ dst, int n) {
    int i = blockIdx.x * blockDim.x + threadIdx.x;
    if (i < n) dst[i] = a[i] + b[i] + (c ? c[i] : 0.0f);
}

// ---------------- persistent recurrent scan ----------------
__global__ void __launch_bounds__(1024)
lstm_scan(const __bf16* __restrict__ xb,                          // [B][T][I] bf16
          const __bf16* __restrict__ bpI, const __bf16* __restrict__ bpF,
          const __bf16* __restrict__ bpG, const __bf16* __restrict__ bpO,  // packed gate W
          const __bf16* __restrict__ bpP,                         // packed Wph
          const float*  __restrict__ gbias,                       // [4][H] fused gate bias
          const float*  __restrict__ pbias,                       // [O] fused proj bias
          float* __restrict__ out)                                // [B][T][O]
{
    __shared__ __align__(16) __bf16 sx[16 * ROWX];      // x_t tile (single buffer)
    __shared__ __align__(16) __bf16 sh[2][16 * ROWH];   // h double buffer (bf16)
    __shared__ __align__(16) float  sp[16 * O_];        // projection logits

    const int tid  = threadIdx.x;
    const int w    = tid >> 5;
    const int lane = tid & 31;
    const int b0   = blockIdx.x * 16;
    const int nl   = (lane & 15);         // N within tile (C/D layout: N = lane&15)
    const int nlane = w * 16 + nl;        // gate output column owned by this lane
    const int mrow = (lane & 16) ? 8 : 0; // C/D layout row offset

    // zero both h buffers (h_{-1} = 0)
    {
        unsigned* z = (unsigned*)&sh[0][0];
        const int nz = (2 * 16 * ROWH * 2) / 4;
        for (int i = tid; i < nz; i += 1024) z[i] = 0u;
    }

    // hoisted per-lane biases (scalars only; added AFTER the GEMMs)
    const float bI = gbias[0 * H_ + nlane];
    const float bF = gbias[1 * H_ + nlane];
    const float bG = gbias[2 * H_ + nlane];
    const float bO = gbias[3 * H_ + nlane];
    const float bP = (w < 16) ? pbias[w * 16 + nl] : 0.0f;

    // packed-B per-lane base addresses (k-block = +1024B imm offset)
    const uintptr_t aI0 = (uintptr_t)(bpI + (size_t)w * KBG * 512 + lane * 8);
    const uintptr_t aF0 = (uintptr_t)(bpF + (size_t)w * KBG * 512 + lane * 8);
    const uintptr_t aG0 = (uintptr_t)(bpG + (size_t)w * KBG * 512 + lane * 8);
    const uintptr_t aO0 = (uintptr_t)(bpO + (size_t)w * KBG * 512 + lane * 8);
    const uintptr_t aP0 = (uintptr_t)(bpP + (size_t)w * KBH * 512 + lane * 8);

    // phase-0 copy coordinates
    const int cm  = tid >> 6;          // 0..15
    const int ck0 = (tid & 63) * 4;    // 0..252

    v8f c;
    #pragma unroll
    for (int r = 0; r < 8; ++r) c[r] = 0.0f;

    __syncthreads();

    for (int t = 0; t < T_; ++t) {
        const __bf16* hcur = sh[t & 1];
        __bf16*       hnxt = sh[(t & 1) ^ 1];

        // ---- phase 0: stage x_t tile into LDS; prefetch x_{t+1} ----
        {
            const uint2 v = *(const uint2*)(xb + ((size_t)(b0 + cm) * T_ + t) * I_ + ck0);
            *(uint2*)(sx + cm * ROWX + ck0) = v;
            if ((tid & 7) == 0 && t + 1 < T_)   // one prefetch per 64B line
                __builtin_prefetch(xb + ((size_t)(b0 + cm) * T_ + (t + 1)) * I_ + ck0, 0, 1);
        }
        __syncthreads();

        // ---- phase 1: fused gate GEMMs  (K = 768 = x|h), all 32 waves ----
        v8f ai = {}, af = {}, ag = {}, ao = {};   // zero-init (inline-0 C)

        // fresh, opaque global addresses each iteration -> loads stay in-loop
        uintptr_t aI = aI0, aF = aF0, aG = aG0, aO = aO0;
        LAUNDER_ADDR(aI); LAUNDER_ADDR(aF); LAUNDER_ADDR(aG); LAUNDER_ADDR(aO);

        #pragma unroll
        for (int kb = 0; kb < KBX; ++kb) {                 // x part (K = 256)
            v16bf a  = load_afrag(sx, ROWX, kb);
            v16bf Bi = load_bfrag(aI, kb * 1024);
            v16bf Bf = load_bfrag(aF, kb * 1024);
            v16bf Bg = load_bfrag(aG, kb * 1024);
            v16bf Bo = load_bfrag(aO, kb * 1024);
            ai = WMMA_BF16(a, Bi, ai);
            af = WMMA_BF16(a, Bf, af);
            ag = WMMA_BF16(a, Bg, ag);
            ao = WMMA_BF16(a, Bo, ao);
        }
        #pragma unroll
        for (int kb = 0; kb < KBH; ++kb) {                 // h part (K = 512)
            v16bf a  = load_afrag(hcur, ROWH, kb);
            v16bf Bi = load_bfrag(aI, (KBX + kb) * 1024);
            v16bf Bf = load_bfrag(aF, (KBX + kb) * 1024);
            v16bf Bg = load_bfrag(aG, (KBX + kb) * 1024);
            v16bf Bo = load_bfrag(aO, (KBX + kb) * 1024);
            ai = WMMA_BF16(a, Bi, ai);
            af = WMMA_BF16(a, Bf, af);
            ag = WMMA_BF16(a, Bg, ag);
            ao = WMMA_BF16(a, Bo, ao);
        }

        // bias + activations + cell/hidden update (c stays in registers)
        #pragma unroll
        for (int r = 0; r < 8; ++r) {
            float iv = sigf(ai[r] + bI);
            float fv = sigf(af[r] + bF);
            float gv = tanf_(ag[r] + bG);
            float ov = sigf(ao[r] + bO);
            float cv = gv * iv + c[r] * fv;
            c[r] = cv;
            float hv = tanf_(cv) * ov;
            hnxt[(r + mrow) * ROWH + nlane] = (__bf16)hv;
        }
        __syncthreads();

        // ---- phase 2: output projection p = h_t @ Wph^T + b (waves 0..15) ----
        if (w < 16) {
            v8f ap = {};
            uintptr_t aP = aP0;
            LAUNDER_ADDR(aP);
            #pragma unroll
            for (int kb = 0; kb < KBH; ++kb) {
                v16bf a  = load_afrag(hnxt, ROWH, kb);
                v16bf Bp = load_bfrag(aP, kb * 1024);
                ap = WMMA_BF16(a, Bp, ap);
            }
            const int ol = w * 16 + nl;
            #pragma unroll
            for (int r = 0; r < 8; ++r) sp[(r + mrow) * O_ + ol] = ap[r] + bP;
        }
        __syncthreads();

        // ---- phase 3: row softmax (wave r handles batch row r), write y ----
        if (w < 16) {
            float v[8];
            float mx = -3.402823466e38f;
            #pragma unroll
            for (int j = 0; j < 8; ++j) { v[j] = sp[w * O_ + lane + 32 * j]; mx = fmaxf(mx, v[j]); }
            for (int s = 16; s; s >>= 1) mx = fmaxf(mx, __shfl_xor(mx, s, 32));
            float sum = 0.0f;
            #pragma unroll
            for (int j = 0; j < 8; ++j) { v[j] = __expf(v[j] - mx); sum += v[j]; }
            for (int s = 16; s; s >>= 1) sum += __shfl_xor(sum, s, 32);
            const float inv = 1.0f / sum;
            float* orow = out + ((size_t)(b0 + w) * T_ + t) * O_;
            #pragma unroll
            for (int j = 0; j < 8; ++j) orow[lane + 32 * j] = v[j] * inv;
        }
        // no barrier needed here: next phase-0 writes sx only; sp reuse is
        // fenced by the next iteration's phase-0/phase-1 barriers.
    }
}

// ---------------- host launcher ----------------
extern "C" void kernel_launch(void* const* d_in, const int* in_sizes, int n_in,
                              void* d_out, int out_size, void* d_ws, size_t ws_size,
                              hipStream_t stream) {
    const float* x     = (const float*)d_in[0];
    const float* Wix_w = (const float*)d_in[1];
    const float* Wix_b = (const float*)d_in[2];
    const float* Wih_w = (const float*)d_in[3];
    const float* Wih_b = (const float*)d_in[4];
    const float* b_i   = (const float*)d_in[5];
    const float* Wfx_w = (const float*)d_in[6];
    const float* Wfx_b = (const float*)d_in[7];
    const float* Wfh_w = (const float*)d_in[8];
    const float* Wfh_b = (const float*)d_in[9];
    const float* b_f   = (const float*)d_in[10];
    const float* Wgx_w = (const float*)d_in[11];
    const float* Wgx_b = (const float*)d_in[12];
    const float* Wgh_w = (const float*)d_in[13];
    const float* Wgh_b = (const float*)d_in[14];
    const float* b_g   = (const float*)d_in[15];
    const float* Wox_w = (const float*)d_in[16];
    const float* Wox_b = (const float*)d_in[17];
    const float* Woh_w = (const float*)d_in[18];
    const float* Woh_b = (const float*)d_in[19];
    const float* b_o   = (const float*)d_in[20];
    const float* Wph_w = (const float*)d_in[21];
    const float* Wph_b = (const float*)d_in[22];
    const float* b_p   = (const float*)d_in[23];

    char* ws = (char*)d_ws;
    size_t off = 0;
    auto alloc = [&](size_t bytes) { char* p = ws + off; off += (bytes + 255) & ~size_t(255); return p; };

    __bf16* xb  = (__bf16*)alloc((size_t)B_ * T_ * I_ * 2);   // 33.5 MB
    __bf16* bpI = (__bf16*)alloc((size_t)H_ * KTOT * 2);      // 768 KB each
    __bf16* bpF = (__bf16*)alloc((size_t)H_ * KTOT * 2);
    __bf16* bpG = (__bf16*)alloc((size_t)H_ * KTOT * 2);
    __bf16* bpO = (__bf16*)alloc((size_t)H_ * KTOT * 2);
    __bf16* bpP = (__bf16*)alloc((size_t)O_ * H_ * 2);        // 256 KB
    float*  gbias = (float*)alloc(4 * H_ * 4);
    float*  pbias = (float*)alloc(O_ * 4);

    // 1) one-time conversion / packing (all independent, stream-ordered)
    {
        int n = B_ * T_ * I_;
        k_cvt_x<<<(n + 255) / 256, 256, 0, stream>>>(x, xb, n);
    }
    {
        int n = H_ * KTOT;
        dim3 g((n + 255) / 256);
        k_pack_gate<<<g, 256, 0, stream>>>(Wix_w, Wih_w, bpI);
        k_pack_gate<<<g, 256, 0, stream>>>(Wfx_w, Wfh_w, bpF);
        k_pack_gate<<<g, 256, 0, stream>>>(Wgx_w, Wgh_w, bpG);
        k_pack_gate<<<g, 256, 0, stream>>>(Wox_w, Woh_w, bpO);
    }
    {
        int n = O_ * H_;
        k_pack_proj<<<(n + 255) / 256, 256, 0, stream>>>(Wph_w, bpP);
    }
    k_add3<<<(H_ + 255) / 256, 256, 0, stream>>>(Wix_b, Wih_b, b_i, gbias + 0 * H_, H_);
    k_add3<<<(H_ + 255) / 256, 256, 0, stream>>>(Wfx_b, Wfh_b, b_f, gbias + 1 * H_, H_);
    k_add3<<<(H_ + 255) / 256, 256, 0, stream>>>(Wgx_b, Wgh_b, b_g, gbias + 2 * H_, H_);
    k_add3<<<(H_ + 255) / 256, 256, 0, stream>>>(Wox_b, Woh_b, b_o, gbias + 3 * H_, H_);
    k_add3<<<(O_ + 255) / 256, 256, 0, stream>>>(Wph_b, b_p, nullptr, pbias, O_);

    // 2) persistent recurrent scan: 8 batch-slices x 32 waves
    lstm_scan<<<dim3(B_ / 16), dim3(1024), 0, stream>>>(
        xb, bpI, bpF, bpG, bpO, bpP, gbias, pbias, (float*)d_out);
}
